// TransformerMIL_54528904790356
// MI455X (gfx1250) — matrix operand order repaired and checked by reading, not compiled, not run on hardware
//
#include <hip/hip_runtime.h>
#include <hip/hip_bf16.h>

// ---------------------------------------------------------------------------
// TransformerMIL forward for MI455X (gfx1250, wave32, WMMA).
// All GEMMs run through v_wmma_f32_16x16x32_bf16 (f32 -> bf16 inputs, f32 acc).
// NOTE: this GEMM requires M%64==0, N%64==0, K%32==0, lda/ldb%4==0 — true for
// every matmul in this model (shapes listed at each call site).
// ---------------------------------------------------------------------------

typedef __bf16 bf16_t;
typedef __attribute__((ext_vector_type(16))) __bf16 v16bf;
typedef __attribute__((ext_vector_type(8)))  __bf16 v8bf;
typedef __attribute__((ext_vector_type(8)))  float  v8f;
typedef __attribute__((ext_vector_type(4)))  float  v4f;

#define F_BIAS      1
#define F_RELU      2
#define F_QSCALE    4
#define F_BTRANS    8
#define F_ACCUM     16
#define F_REMAP_FC1 32
#define F_REMAP_OUT 64

// ---------------- generic batched bf16 WMMA GEMM -----------------------------
// C[r][c] = sum_k A[r][k] * B[k][c]   (B given as [N][K] when BTRANS)
// block tile 64x64, K step 32, 256 threads = 8 waves, each wave two 16x16 tiles
template <bool BTRANS>
__global__ __launch_bounds__(256) void gemm_bf16_wmma(
    const float* __restrict__ A, const float* __restrict__ Bm,
    const float* __restrict__ bias, float* __restrict__ Cm,
    int M, int N, int K, int lda, int ldb, int ldc,
    long long sAb, long long sAh, long long sBb, long long sBh,
    long long sCb, long long sCh, int innerH,
    int flags, float qscale, int qcols, float dmul, float dadd,
    int pad, int seg_in, int seg_out)
{
  __shared__ __align__(16) bf16_t Als[64][32];   // [m][k]
  __shared__ __align__(16) bf16_t Bls[64][32];   // [n][k] (transposed stage)
  {
    int z = blockIdx.z;
    int bo = z / innerH, hi = z - bo * innerH;
    A  += bo * sAb + hi * sAh;
    Bm += bo * sBb + hi * sBh;
    Cm += bo * sCb + hi * sCh;
  }
  const int bm = blockIdx.y, bn = blockIdx.x;
  const int tid  = threadIdx.x;
  const int lane = tid & 31, wave = tid >> 5;
  const int tm  = wave & 3;          // M tile 0..3
  const int tn0 = (wave >> 2) * 2;   // N tiles {tn0, tn0+1}
  const int lrow = lane & 15, lhi = lane >> 4;
  const int ldr = tid >> 2;          // 0..63 (row of tile to load)
  const int ldk = (tid & 3) * 8;     // k sub-chunk

  // running source pointers (strength-reduced: advance per K step)
  const float* aPtr = A + (long long)(bm * 64 + ldr) * lda + ldk;
  const float* bPtr = BTRANS ? (Bm + (long long)(bn * 64 + ldr) * ldb + ldk)
                             : (Bm + (long long)ldk * ldb + (bn * 64 + ldr));
  const long long bStep = BTRANS ? 32 : (long long)32 * ldb;

  v8f c0 = {};
  v8f c1 = {};

  for (int k0 = 0; k0 < K; k0 += 32) {
    { // A: 8 contiguous f32 -> 8 bf16, single 16B LDS store
      v4f f0 = ((const v4f*)aPtr)[0], f1 = ((const v4f*)aPtr)[1];
      v8bf pk;
      #pragma unroll
      for (int j = 0; j < 4; ++j) { pk[j] = (bf16_t)f0[j]; pk[4 + j] = (bf16_t)f1[j]; }
      *(v8bf*)&Als[ldr][ldk] = pk;
    }
    if (BTRANS) { // B as [N][K]: contiguous along K
      v4f f0 = ((const v4f*)bPtr)[0], f1 = ((const v4f*)bPtr)[1];
      v8bf pk;
      #pragma unroll
      for (int j = 0; j < 4; ++j) { pk[j] = (bf16_t)f0[j]; pk[4 + j] = (bf16_t)f1[j]; }
      *(v8bf*)&Bls[ldr][ldk] = pk;
    } else {      // B as [K][N]: 8 loads down column, base + immediate offsets
      v8bf pk;
      #pragma unroll
      for (int j = 0; j < 8; ++j) pk[j] = (bf16_t)bPtr[(long long)j * ldb];
      *(v8bf*)&Bls[ldr][ldk] = pk;
    }
    aPtr += 32;
    bPtr += bStep;
    // prefetch next K tile into cache (global_prefetch_b8)
    if (k0 + 32 < K) {
      __builtin_prefetch(aPtr, 0, 1);
      __builtin_prefetch(bPtr, 0, 1);
    }
    __syncthreads();
    // A fragment (16x32, 16-bit): lane=row(l&15), K = (l>>4)*16 + 0..15 contiguous
    v8bf alo = *(const v8bf*)&Als[tm * 16 + lrow][lhi * 16];
    v8bf ahi = *(const v8bf*)&Als[tm * 16 + lrow][lhi * 16 + 8];
    // B fragment (32x16): lane=col(l&15), K = (l>>4)*16 + 0..15 contiguous
    v8bf b0l = *(const v8bf*)&Bls[tn0 * 16 + lrow][lhi * 16];
    v8bf b0h = *(const v8bf*)&Bls[tn0 * 16 + lrow][lhi * 16 + 8];
    v8bf b1l = *(const v8bf*)&Bls[tn0 * 16 + 16 + lrow][lhi * 16];
    v8bf b1h = *(const v8bf*)&Bls[tn0 * 16 + 16 + lrow][lhi * 16 + 8];
    v16bf af, bf0, bf1;
    #pragma unroll
    for (int i = 0; i < 8; ++i) {
      af[i]  = alo[i]; af[i + 8]  = ahi[i];
      bf0[i] = b0l[i]; bf0[i + 8] = b0h[i];
      bf1[i] = b1l[i]; bf1[i + 8] = b1h[i];
    }
    c0 = __builtin_amdgcn_wmma_f32_16x16x32_bf16(false, af, false, bf0,
                                                 (short)0, c0, false, false);
    c1 = __builtin_amdgcn_wmma_f32_16x16x32_bf16(false, af, false, bf1,
                                                 (short)0, c1, false, false);
    __syncthreads();
  }

  // epilogue: C/D layout: VGPR j -> row (l>>4)*8 + j, col = l&15
  #pragma unroll
  for (int half = 0; half < 2; ++half) {
    v8f acc = half ? c1 : c0;
    int tn = tn0 + half;
    #pragma unroll
    for (int j = 0; j < 8; ++j) {
      int r = bm * 64 + tm * 16 + lhi * 8 + j;
      int c = bn * 64 + tn * 16 + lrow;
      if (r >= M || c >= N) continue;
      float v = acc[j];
      if (flags & F_BIAS)   v += bias[c];
      if (flags & F_QSCALE) { if (c < qcols) v *= qscale; }
      if (flags & F_RELU)   v = fmaxf(v, 0.f);
      v = dmul * v + ((r == c) ? dadd : 0.f);
      long long drow = r;
      if (flags & F_REMAP_FC1) drow = (long long)r + r / seg_in + 1;
      if (flags & F_REMAP_OUT) {
        int bI = r / seg_in, ii = r - bI * seg_in;
        if (ii < pad) continue;
        drow = (long long)bI * seg_out + (ii - pad);
      }
      float* dst = Cm + drow * (long long)ldc + c;
      if (flags & F_ACCUM) *dst += v; else *dst = v;
    }
  }
}

// ---------------- wave32 helpers --------------------------------------------
__device__ __forceinline__ float wave_sum(float v) {
  #pragma unroll
  for (int o = 16; o; o >>= 1) v += __shfl_xor(v, o, 32);
  return v;
}
__device__ __forceinline__ float wave_max(float v) {
  #pragma unroll
  for (int o = 16; o; o >>= 1) v = fmaxf(v, __shfl_xor(v, o, 32));
  return v;
}

// write cls token into h[b][0][:]
__global__ void cls_kernel(const float* __restrict__ cls, float* __restrict__ h) {
  int c = blockIdx.x * 256 + threadIdx.x;
  if (c < 512) {
    h[c] = cls[c];
    h[(long long)8001 * 512 + c] = cls[c];
  }
}

// layernorm h[b][i-191] -> xp[b][i], rows i<191 zeroed (front pad). wave per row.
__global__ void lnpad_kernel(const float* __restrict__ h, const float* __restrict__ g,
                             const float* __restrict__ be, float* __restrict__ xp) {
  int row  = blockIdx.x * 8 + (threadIdx.x >> 5);   // 0..16383
  int lane = threadIdx.x & 31;
  int b = row >> 13, i = row & 8191;
  float* dst = xp + (long long)row * 512;
  if (i < 191) {
    #pragma unroll
    for (int t = 0; t < 16; ++t) dst[lane + t * 32] = 0.f;
    return;
  }
  const float* src = h + ((long long)b * 8001 + (i - 191)) * 512;
  float x[16]; float s = 0.f, ss = 0.f;
  #pragma unroll
  for (int t = 0; t < 16; ++t) { float v = src[lane + t * 32]; x[t] = v; s += v; ss += v * v; }
  s = wave_sum(s); ss = wave_sum(ss);
  float mu = s * (1.f / 512.f);
  float var = ss * (1.f / 512.f) - mu * mu;
  float rs = rsqrtf(var + 1e-5f);
  #pragma unroll
  for (int t = 0; t < 16; ++t) {
    int c = lane + t * 32;
    dst[c] = (x[t] - mu) * rs * g[c] + be[c];
  }
}

// landmark means: ql/kl[b][h][m][d] = mean over 32 tokens. block=(64), grid=b*h*m
__global__ void landmark_kernel(const float* __restrict__ qkv,
                                float* __restrict__ ql, float* __restrict__ kl) {
  int idx = blockIdx.x;                  // b*2048 + h*256 + m
  int m = idx & 255, hh = (idx >> 8) & 7, b = idx >> 11;
  int d = threadIdx.x;
  const float* base = qkv + ((long long)b * 8192 + m * 32) * 1536 + hh * 64 + d;
  float sq = 0.f, sk = 0.f;
  for (int l = 0; l < 32; ++l) {
    sq += base[(long long)l * 1536];
    sk += base[(long long)l * 1536 + 512];
  }
  long long o = (long long)idx * 64 + d;
  ql[o] = sq * (1.f / 32.f);
  kl[o] = sk * (1.f / 32.f);
}

// row softmax, 256 cols, one wave per row
__global__ void softmax256_kernel(float* __restrict__ X, long long rows) {
  long long row = (long long)blockIdx.x * 8 + (threadIdx.x >> 5);
  int lane = threadIdx.x & 31;
  if (row >= rows) return;
  float* p = X + row * 256;
  float v[8]; float mx = -3.4e38f;
  #pragma unroll
  for (int i = 0; i < 8; ++i) { v[i] = p[lane + i * 32]; mx = fmaxf(mx, v[i]); }
  mx = wave_max(mx);
  float s = 0.f;
  #pragma unroll
  for (int i = 0; i < 8; ++i) { v[i] = __expf(v[i] - mx); s += v[i]; }
  s = wave_sum(s);
  float inv = 1.f / s;
  #pragma unroll
  for (int i = 0; i < 8; ++i) p[lane + i * 32] = v[i] * inv;
}

// row softmax, wide (8192 cols), one block (256 thr) per row
__global__ void softmax_wide_kernel(float* __restrict__ X, int cols) {
  long long row = blockIdx.x;
  float* p = X + row * (long long)cols;
  __shared__ float red[8];
  int tid = threadIdx.x, lane = tid & 31, w = tid >> 5;
  float mx = -3.4e38f;
  for (int c = tid; c < cols; c += 256) mx = fmaxf(mx, p[c]);
  mx = wave_max(mx);
  if (lane == 0) red[w] = mx;
  __syncthreads();
  mx = wave_max((lane < 8) ? red[lane] : -3.4e38f);
  float s = 0.f;
  for (int c = tid; c < cols; c += 256) s += __expf(p[c] - mx);
  s = wave_sum(s);
  __syncthreads();
  if (lane == 0) red[w] = s;
  __syncthreads();
  s = wave_sum((lane < 8) ? red[lane] : 0.f);
  float inv = 1.f / s;
  for (int c = tid; c < cols; c += 256) p[c] = __expf(p[c] - mx) * inv;
}

__global__ void zero2_kernel(float* __restrict__ p) {
  if (threadIdx.x < 2) p[threadIdx.x] = 0.f;
}

// max over rows of |row|.sum  (a2 is 16*256 rows of 256)
__global__ void rowsummax_kernel(const float* __restrict__ a2, float* __restrict__ out) {
  long long row = blockIdx.x;
  int tid = threadIdx.x, lane = tid & 31, w = tid >> 5;
  __shared__ float red[8];
  float s = wave_sum(fabsf(a2[row * 256 + tid]));
  if (lane == 0) red[w] = s;
  __syncthreads();
  if (tid < 32) {
    float t = wave_sum((tid < 8) ? red[tid] : 0.f);
    if (tid == 0) atomicMax((int*)out, __float_as_int(t));
  }
}
// max over cols of |col|.sum
__global__ void colsummax_kernel(const float* __restrict__ a2, float* __restrict__ out) {
  int bh = blockIdx.x >> 8, col = blockIdx.x & 255;
  int tid = threadIdx.x, lane = tid & 31, w = tid >> 5;
  __shared__ float red[8];
  float s = wave_sum(fabsf(a2[((long long)bh * 256 + tid) * 256 + col]));
  if (lane == 0) red[w] = s;
  __syncthreads();
  if (tid < 32) {
    float t = wave_sum((tid < 8) ? red[tid] : 0.f);
    if (tid == 0) atomicMax((int*)out, __float_as_int(t));
  }
}

// z[bh][i][j] = a2[bh][j][i] / (sc0*sc1)
__global__ void zinit_kernel(const float* __restrict__ a2, float* __restrict__ z,
                             const float* __restrict__ sc) {
  float inv = 1.f / (sc[0] * sc[1]);
  long long t = (long long)blockIdx.x * 256 + threadIdx.x;   // 16*65536 total
  int j = t & 255; int i = (int)((t >> 8) & 255); long long bh = t >> 16;
  z[t] = a2[(bh << 16) + (long long)j * 256 + i] * inv;
}

// out = alpha*I - in  over [16][256][256]
__global__ void diagsub_kernel(float* __restrict__ out, const float* __restrict__ in,
                               float alpha) {
  long long t = (long long)blockIdx.x * 256 + threadIdx.x;
  int j = t & 255; int i = (int)((t >> 8) & 255);
  out[t] = ((i == j) ? alpha : 0.f) - in[t];
}

// depthwise conv over sequence on v (inside qkv), K=33, pad 16 -> attn buffer
__global__ void dwconv_kernel(const float* __restrict__ qkv, const float* __restrict__ w,
                              float* __restrict__ attn) {
  long long t = (long long)blockIdx.x * 256 + threadIdx.x;  // B*8192*512
  int c = t & 511; long long bi = t >> 9; int i = (int)(bi & 8191); int b = (int)(bi >> 13);
  int hh = c >> 6, d = c & 63;
  const float* vb = qkv + ((long long)b * 8192) * 1536 + 1024 + hh * 64 + d;
  float s = 0.f;
  #pragma unroll
  for (int tt = 0; tt < 33; ++tt) {
    int ii = i + tt - 16;
    if (ii >= 0 && ii < 8192) s += vb[(long long)ii * 1536] * w[hh * 33 + tt];
  }
  attn[t] = s;
}

// final head: layernorm h[b][0], fc2, softmax, argmax. grid=B blocks, 32 threads.
__global__ void head_kernel(const float* __restrict__ h, const float* __restrict__ g,
                            const float* __restrict__ be, const float* __restrict__ w2,
                            const float* __restrict__ b2, float* __restrict__ out) {
  int b = blockIdx.x;
  int lane = threadIdx.x;
  const float* row = h + (long long)b * 8001 * 512;
  float x[16]; float s = 0.f, ss = 0.f;
  #pragma unroll
  for (int t = 0; t < 16; ++t) { float v = row[lane + t * 32]; x[t] = v; s += v; ss += v * v; }
  s = wave_sum(s); ss = wave_sum(ss);
  float mu = s * (1.f / 512.f);
  float var = ss * (1.f / 512.f) - mu * mu;
  float rs = rsqrtf(var + 1e-5f);
  float d0 = 0.f, d1 = 0.f;
  #pragma unroll
  for (int t = 0; t < 16; ++t) {
    int c = lane + t * 32;
    float nv = (x[t] - mu) * rs * g[c] + be[c];
    d0 += nv * w2[c * 2 + 0];
    d1 += nv * w2[c * 2 + 1];
  }
  d0 = wave_sum(d0); d1 = wave_sum(d1);
  if (lane == 0) {
    float l0 = d0 + b2[0], l1 = d1 + b2[1];
    float m = fmaxf(l0, l1);
    float e0 = __expf(l0 - m), e1 = __expf(l1 - m);
    float inv = 1.f / (e0 + e1);
    out[b * 2 + 0] = l0;           out[b * 2 + 1] = l1;       // logits
    out[4 + b * 2 + 0] = e0 * inv; out[4 + b * 2 + 1] = e1 * inv; // probs
    out[8 + b] = (l1 > l0) ? 1.f : 0.f;                       // argmax
  }
}

// ---------------- host-side orchestration -----------------------------------
static inline void gemm(hipStream_t s,
    const float* A, const float* B, const float* bias, float* C,
    int M, int N, int K, int lda, int ldb, int ldc,
    long long sAb, long long sAh, long long sBb, long long sBh,
    long long sCb, long long sCh, int innerH, int nbatch,
    int flags, float qscale = 0.f, int qcols = 0,
    float dmul = 1.f, float dadd = 0.f,
    int pad = 0, int seg_in = 1, int seg_out = 0) {
  dim3 grid((N + 63) / 64, (M + 63) / 64, nbatch);
  if (flags & F_BTRANS)
    gemm_bf16_wmma<true><<<grid, 256, 0, s>>>(A, B, bias, C, M, N, K, lda, ldb,
        ldc, sAb, sAh, sBb, sBh, sCb, sCh, innerH, flags, qscale, qcols, dmul,
        dadd, pad, seg_in, seg_out);
  else
    gemm_bf16_wmma<false><<<grid, 256, 0, s>>>(A, B, bias, C, M, N, K, lda, ldb,
        ldc, sAb, sAh, sBb, sBh, sCb, sCh, innerH, flags, qscale, qcols, dmul,
        dadd, pad, seg_in, seg_out);
}

// workspace layout (floats)
static const long long SZ_H    = 2LL * 8001 * 512;
static const long long SZ_XP   = 2LL * 8192 * 512;
static const long long SZ_QKV  = 2LL * 8192 * 1536;
static const long long SZ_QL   = 16LL * 256 * 64;
static const long long SZ_A1   = 16LL * 8192 * 256;
static const long long SZ_A2   = 16LL * 256 * 256;
static const long long SZ_A3   = 16LL * 256 * 8192;   // also reused for t = a1@z
static const long long SZ_A3V  = 16LL * 256 * 64;
static const long long SZ_ATTN = 2LL * 8192 * 512;

static const long long O_H    = 0;
static const long long O_XP   = O_H    + SZ_H;
static const long long O_QKV  = O_XP   + SZ_XP;
static const long long O_QL   = O_QKV  + SZ_QKV;
static const long long O_KL   = O_QL   + SZ_QL;
static const long long O_A1   = O_KL   + SZ_QL;
static const long long O_A2   = O_A1   + SZ_A1;
static const long long O_XZ   = O_A2   + SZ_A2;
static const long long O_W1   = O_XZ   + SZ_A2;
static const long long O_W2   = O_W1   + SZ_A2;
static const long long O_Z0   = O_W2   + SZ_A2;
static const long long O_Z1   = O_Z0   + SZ_A2;
static const long long O_A3   = O_Z1   + SZ_A2;
static const long long O_A3V  = O_A3   + SZ_A3;
static const long long O_ATTN = O_A3V  + SZ_A3V;
static const long long O_SC   = O_ATTN + SZ_ATTN;

static void run_layer(hipStream_t s, float* ws,
                      const float* lng, const float* lnb, const float* qkvw,
                      const float* outw, const float* outb, const float* convw) {
  float* H_  = ws + O_H;
  float* XP  = ws + O_XP;
  float* QKV = ws + O_QKV;
  float* QL  = ws + O_QL;
  float* KL  = ws + O_KL;
  float* A1  = ws + O_A1;
  float* A2  = ws + O_A2;
  float* XZ  = ws + O_XZ;
  float* W1  = ws + O_W1;
  float* W2  = ws + O_W2;
  float* Z0  = ws + O_Z0;
  float* Z1  = ws + O_Z1;
  float* A3  = ws + O_A3;   // later reused as t = a1@z
  float* A3V = ws + O_A3V;
  float* ATT = ws + O_ATTN;
  float* SC  = ws + O_SC;

  const long long sQb = 8192LL * 1536, sQh = 64;           // q/k/v views in qkv
  const long long sLb = 8LL * 256 * 64, sLh = 256LL * 64;  // ql/kl
  const long long sMb = 8LL * 256 * 256, sMh = 256LL * 256;// 256x256 mats

  // layernorm + front pad (191 zero rows)
  lnpad_kernel<<<2048, 256, 0, s>>>(H_, lng, lnb, XP);
  // qkv = xp @ qkv_w ; fold q *= dh^-0.5 into cols < 512   [16384 x 1536 x 512]
  gemm(s, XP, qkvw, nullptr, QKV, 16384, 1536, 512, 512, 1536, 1536,
       0, 0, 0, 0, 0, 0, 1, 1, F_QSCALE, 0.125f, 512);
  // landmark means
  landmark_kernel<<<4096, 64, 0, s>>>(QKV, QL, KL);
  // a1 = softmax(q @ kl^T)  [16 x 8192 x 256, K=64]
  gemm(s, QKV, KL, nullptr, A1, 8192, 256, 64, 1536, 64, 256,
       sQb, sQh, sLb, sLh, 8LL * 8192 * 256, 8192LL * 256, 8, 16, F_BTRANS);
  softmax256_kernel<<<16384, 256, 0, s>>>(A1, 16LL * 8192);
  // a2 = softmax(ql @ kl^T)  [16 x 256 x 256, K=64]
  gemm(s, QL, KL, nullptr, A2, 256, 256, 64, 64, 64, 256,
       sLb, sLh, sLb, sLh, sMb, sMh, 8, 16, F_BTRANS);
  softmax256_kernel<<<512, 256, 0, s>>>(A2, 4096);
  // pinv denom = max rowsum * max colsum (global)
  zero2_kernel<<<1, 32, 0, s>>>(SC);
  rowsummax_kernel<<<4096, 256, 0, s>>>(A2, SC);
  colsummax_kernel<<<4096, 256, 0, s>>>(A2, SC + 1);
  zinit_kernel<<<4096, 256, 0, s>>>(A2, Z0, SC);
  // 6 Newton-Schulz style iterations (batched 16 x 256^3)
  float* zc = Z0; float* zn = Z1;
  for (int it = 0; it < 6; ++it) {
    gemm(s, A2, zc, nullptr, XZ, 256, 256, 256, 256, 256, 256,
         sMb, sMh, sMb, sMh, sMb, sMh, 8, 16, 0);
    diagsub_kernel<<<4096, 256, 0, s>>>(W1, XZ, 7.f);                 // W1 = 7I - xz
    gemm(s, XZ, W1, nullptr, W2, 256, 256, 256, 256, 256, 256,
         sMb, sMh, sMb, sMh, sMb, sMh, 8, 16, 0, 0.f, 0, -1.f, 15.f); // W2 = 15I - xz@W1
    gemm(s, XZ, W2, nullptr, W1, 256, 256, 256, 256, 256, 256,
         sMb, sMh, sMb, sMh, sMb, sMh, 8, 16, 0, 0.f, 0, -1.f, 13.f); // W1 = 13I - xz@W2
    gemm(s, zc, W1, nullptr, zn, 256, 256, 256, 256, 256, 256,
         sMb, sMh, sMb, sMh, sMb, sMh, 8, 16, 0, 0.f, 0, 0.25f, 0.f); // zn = 0.25*z@W1
    float* tmp = zc; zc = zn; zn = tmp;
  }
  // a3 = softmax(ql @ k^T)  [16 x 256 x 8192, K=64]
  gemm(s, QL, QKV + 512, nullptr, A3, 256, 8192, 64, 64, 1536, 8192,
       sLb, sLh, sQb, sQh, 8LL * 256 * 8192, 256LL * 8192, 8, 16, F_BTRANS);
  softmax_wide_kernel<<<4096, 256, 0, s>>>(A3, 8192);
  // a3v = a3 @ v  [16 x 256 x 64, K=8192]
  gemm(s, A3, QKV + 1024, nullptr, A3V, 256, 64, 8192, 8192, 1536, 64,
       8LL * 256 * 8192, 256LL * 8192, sQb, sQh, sLb, sLh, 8, 16, 0);
  // depthwise conv residual -> ATT
  dwconv_kernel<<<32768, 256, 0, s>>>(QKV, convw, ATT);
  // t = a1 @ z   (reuse A3 buffer)  [16 x 8192 x 256, K=256]
  gemm(s, A1, zc, nullptr, A3, 8192, 256, 256, 256, 256, 256,
       8LL * 8192 * 256, 8192LL * 256, sMb, sMh,
       8LL * 8192 * 256, 8192LL * 256, 8, 16, 0);
  // ATT += t @ a3v  (per head into [B,8192,512])  [16 x 8192 x 64, K=256]
  gemm(s, A3, A3V, nullptr, ATT, 8192, 64, 256, 256, 64, 512,
       8LL * 8192 * 256, 8192LL * 256, sLb, sLh,
       8192LL * 512, 64, 8, 16, F_ACCUM);
  // h += (ATT @ out_w + out_b), dropping 191 front-pad rows  [16384 x 512 x 512]
  gemm(s, ATT, outw, outb, H_, 16384, 512, 512, 512, 512, 512,
       0, 0, 0, 0, 0, 0, 1, 1, F_BIAS | F_ACCUM | F_REMAP_OUT,
       0.f, 0, 1.f, 0.f, 191, 8192, 8001);
}

extern "C" void kernel_launch(void* const* d_in, const int* in_sizes, int n_in,
                              void* d_out, int out_size, void* d_ws, size_t ws_size,
                              hipStream_t stream) {
  (void)in_sizes; (void)n_in; (void)out_size; (void)ws_size;
  const float* feats   = (const float*)d_in[0];
  const float* fc1_w   = (const float*)d_in[1];
  const float* fc1_b   = (const float*)d_in[2];
  const float* cls_t   = (const float*)d_in[3];
  const float* ln1_g   = (const float*)d_in[4];
  const float* ln1_b   = (const float*)d_in[5];
  const float* qkv1_w  = (const float*)d_in[6];
  const float* out1_w  = (const float*)d_in[7];
  const float* out1_b  = (const float*)d_in[8];
  const float* conv1_w = (const float*)d_in[9];
  const float* ln2_g   = (const float*)d_in[10];
  const float* ln2_b   = (const float*)d_in[11];
  const float* qkv2_w  = (const float*)d_in[12];
  const float* out2_w  = (const float*)d_in[13];
  const float* out2_b  = (const float*)d_in[14];
  const float* conv2_w = (const float*)d_in[15];
  const float* normf_g = (const float*)d_in[16];
  const float* normf_b = (const float*)d_in[17];
  const float* fc2_w   = (const float*)d_in[18];
  const float* fc2_b   = (const float*)d_in[19];

  float* ws = (float*)d_ws;
  float* H_ = ws + O_H;

  // h[:,0] = cls token; h[:,1:] = relu(feats @ fc1_w + fc1_b)  [16000 x 512 x 1024]
  cls_kernel<<<2, 256, 0, stream>>>(cls_t, H_);
  gemm(stream, feats, fc1_w, fc1_b, H_, 16000, 512, 1024, 1024, 512, 512,
       0, 0, 0, 0, 0, 0, 1, 1, F_BIAS | F_RELU | F_REMAP_FC1,
       0.f, 0, 1.f, 0.f, 0, 8000, 0);

  run_layer(stream, ws, ln1_g, ln1_b, qkv1_w, out1_w, out1_b, conv1_w);
  run_layer(stream, ws, ln2_g, ln2_b, qkv2_w, out2_w, out2_b, conv2_w);

  head_kernel<<<2, 32, 0, stream>>>(H_, normf_g, normf_b, fc2_w, fc2_b,
                                    (float*)d_out);
}
